// VimBlock_69174743269731
// MI455X (gfx1250) — matrix operand order repaired and checked by reading, not compiled, not run on hardware
//
#include <hip/hip_runtime.h>
#include <hip/hip_bf16.h>

// ---------------------------------------------------------------------------
// Vision-Mamba (bidirectional selective SSM) block, fused for gfx1250.
// All GEMMs via v_wmma_f32_16x16x32_f16 (f16 in, f32 accum).
// Sequential scan -> wave32 chunked parallel scan with shfl composition.
// ---------------------------------------------------------------------------

#define B_ 32
#define J_ 1024
#define D_ 192
#define E_ 384
#define N_ 16
#define M_ (B_ * J_)   // 32768 rows

typedef __attribute__((ext_vector_type(16))) _Float16 v16h;
typedef __attribute__((ext_vector_type(8)))  _Float16 v8h;
typedef __attribute__((ext_vector_type(8)))  float    v8f;

__device__ __forceinline__ float softplusf(float x) {
  return x > 20.f ? x : log1pf(expf(x));
}
__device__ __forceinline__ float sigmoidf(float x) {
  return 1.f / (1.f + expf(-x));
}

// ---- WMMA fragment loaders -------------------------------------------------
// A (16x32 f16, row-major, lda halfs): lane L holds row m=L&15.
//   halves h = K k0 + (hi?16:0) + h  -> 16 contiguous halfs.
__device__ __forceinline__ v16h load_a_frag(const _Float16* __restrict__ A,
                                            int lda, int m0, int k0, int lane) {
  int m  = m0 + (lane & 15);
  int hi = lane >> 4;
  const _Float16* p = A + (size_t)m * lda + k0 + hi * 16;
  v8h lo = *reinterpret_cast<const v8h*>(p);
  v8h hh = *reinterpret_cast<const v8h*>(p + 8);
  return __builtin_shufflevector(lo, hh, 0,1,2,3,4,5,6,7,8,9,10,11,12,13,14,15);
}
// B (32x16 f16) from column-major BT[n*ldb + k]: lane L holds col n=L&15,
//   halves h = K k0 + (hi?16:0) + h  -> 16 contiguous halfs.
__device__ __forceinline__ v16h load_b_frag(const _Float16* __restrict__ BT,
                                            int ldb, int n0, int k0, int lane) {
  int n  = n0 + (lane & 15);
  int hi = lane >> 4;
  const _Float16* p = BT + (size_t)n * ldb + k0 + hi * 16;
  v8h lo = *reinterpret_cast<const v8h*>(p);
  v8h hh = *reinterpret_cast<const v8h*>(p + 8);
  return __builtin_shufflevector(lo, hh, 0,1,2,3,4,5,6,7,8,9,10,11,12,13,14,15);
}

__device__ __forceinline__ v8f wmma_f16(v16h a, v16h b, v8f c) {
  return __builtin_amdgcn_wmma_f32_16x16x32_f16(false, a, false, b,
                                                (short)0, c, false, false);
}

// one wave computes a 16 x (16*NT) strip
template <int NT>
__device__ __forceinline__ void gemm_strip(const _Float16* __restrict__ A, int lda,
                                           const _Float16* __restrict__ BT, int ldb,
                                           int K, int m0, int n0, int lane,
                                           v8f acc[NT]) {
  for (int k0 = 0; k0 < K; k0 += 32) {
    v16h af = load_a_frag(A, lda, m0, k0, lane);
#pragma unroll
    for (int t = 0; t < NT; ++t) {
      v16h bf = load_b_frag(BT, ldb, n0 + t * 16, k0, lane);
      acc[t] = wmma_f16(af, bf, acc[t]);
    }
  }
}

// ---- weight prep (transpose to f16 column-major) ---------------------------
__global__ void prep_wxz(const float* __restrict__ Wx, const float* __restrict__ Wz,
                         _Float16* __restrict__ dst) {  // dst[col*192+k], col<768
  int idx = blockIdx.x * 256 + threadIdx.x;
  if (idx >= 768 * D_) return;
  int col = idx / D_, k = idx % D_;
  float v = (col < E_) ? Wx[(size_t)k * E_ + col] : Wz[(size_t)k * E_ + (col - E_)];
  dst[(size_t)col * D_ + k] = (_Float16)v;
}

__global__ void prep_wcat(const float* __restrict__ WB, const float* __restrict__ WC,
                          const float* __restrict__ WD, const float* __restrict__ Wi,
                          _Float16* __restrict__ dst) {  // dst[col*384+e], col<64
  int idx = blockIdx.x * 256 + threadIdx.x;
  if (idx >= 64 * E_) return;
  int col = idx / E_, e = idx % E_;
  int g = col >> 4, nn = col & 15;
  const float* src = (g == 0) ? WB : (g == 1) ? WC : (g == 2) ? WD : Wi;
  dst[(size_t)col * E_ + e] = (_Float16)src[(size_t)e * N_ + nn];
}

__global__ void prep_wr2(const float* __restrict__ Wr, _Float16* __restrict__ dst) {
  // dst[col*32+kk] = Wr[kk&15][col]  (duplicated for fwd/bwd halves of K=32)
  int idx = blockIdx.x * 256 + threadIdx.x;
  if (idx >= E_ * 32) return;
  int col = idx / 32, kk = idx % 32;
  dst[(size_t)col * 32 + kk] = (_Float16)Wr[(size_t)(kk & 15) * E_ + col];
}

__global__ void prep_woT(const float* __restrict__ Wo, _Float16* __restrict__ dst) {
  int idx = blockIdx.x * 256 + threadIdx.x;
  if (idx >= D_ * E_) return;
  int col = idx / E_, e = idx % E_;
  dst[(size_t)col * E_ + e] = (_Float16)Wo[(size_t)e * D_ + col];
}

// ---- LayerNorm -> f16 ------------------------------------------------------
__global__ void ln_kernel(const float* __restrict__ tok, const float* __restrict__ g,
                          const float* __restrict__ bta, _Float16* __restrict__ t16) {
  int wave = threadIdx.x >> 5, lane = threadIdx.x & 31;
  int row = blockIdx.x * (blockDim.x >> 5) + wave;      // 192 elems / row
  const float* r = tok + (size_t)row * D_;
  float v[6], s = 0.f;
#pragma unroll
  for (int i = 0; i < 6; ++i) { v[i] = r[lane + i * 32]; s += v[i]; }
  for (int off = 16; off; off >>= 1) s += __shfl_xor(s, off, 32);
  float mu = s * (1.f / D_), q = 0.f;
#pragma unroll
  for (int i = 0; i < 6; ++i) { float d = v[i] - mu; q += d * d; }
  for (int off = 16; off; off >>= 1) q += __shfl_xor(q, off, 32);
  float rstd = rsqrtf(q * (1.f / D_) + 1e-5f);
#pragma unroll
  for (int i = 0; i < 6; ++i) {
    int c = lane + i * 32;
    t16[(size_t)row * D_ + c] = (_Float16)((v[i] - mu) * rstd * g[c] + bta[c]);
  }
}

// ---- x/z projection GEMM, fused gate epilogue ------------------------------
__global__ void gemm_xz_kernel(const _Float16* __restrict__ t16,
                               const _Float16* __restrict__ WxzT,
                               const float* __restrict__ bx, const float* __restrict__ bz,
                               _Float16* __restrict__ x16, _Float16* __restrict__ gate16) {
  int lane = threadIdx.x;
  int m0 = blockIdx.x * 16, n0 = blockIdx.y * 64;
  v8f acc[4] = {};
  gemm_strip<4>(t16, D_, WxzT, D_, D_, m0, n0, lane, acc);
  int nc = lane & 15, hi = lane >> 4;
#pragma unroll
  for (int t = 0; t < 4; ++t) {
    int col = n0 + t * 16 + nc;
    if (col < E_) {
      float bb = bx[col];
#pragma unroll
      for (int r = 0; r < 8; ++r) {
        int row = m0 + r + hi * 8;
        x16[(size_t)row * E_ + col] = (_Float16)(acc[t][r] + bb);
      }
    } else {
      int cz = col - E_;
      float bb = bz[cz];
#pragma unroll
      for (int r = 0; r < 8; ++r) {
        int row = m0 + r + hi * 8;
        float z  = acc[t][r] + bb;
        float si = z * sigmoidf(z);               // silu
        gate16[(size_t)row * E_ + cz] = (_Float16)sigmoidf(si);
      }
    }
  }
}

// ---- depthwise conv K=3 (with optional sequence flip on read) --------------
__global__ void dwconv_kernel(const _Float16* __restrict__ x16,
                              const float* __restrict__ cw, const float* __restrict__ cb,
                              _Float16* __restrict__ xc16, int dir) {
  int idx = blockIdx.x * 256 + threadIdx.x;
  int e = idx % E_;
  int s = (idx / E_) & (J_ - 1);
  int b = idx / (E_ * J_);
  float acc = cb[e];
#pragma unroll
  for (int k = 0; k < 3; ++k) {
    int t = s + k - 1;
    if (t >= 0 && t < J_) {
      int j = dir ? (J_ - 1 - t) : t;
      acc += cw[e * 3 + k] * (float)x16[((size_t)b * J_ + j) * E_ + e];
    }
  }
  xc16[((size_t)b * J_ + s) * E_ + e] = (_Float16)acc;
}

// ---- B/C/Delta/u projection GEMM, fused ZOH-discretization epilogue --------
__global__ void gemm_proj_kernel(const _Float16* __restrict__ xc16,
                                 const _Float16* __restrict__ WcatT,
                                 const float* __restrict__ bB, const float* __restrict__ bC,
                                 const float* __restrict__ bD, const float* __restrict__ bi,
                                 const float* __restrict__ A_log,
                                 float* __restrict__ Abar, float* __restrict__ Bu,
                                 float* __restrict__ Ct) {
  int lane = threadIdx.x;
  int m0 = blockIdx.x * 16;
  v8f acc[4] = {};
  gemm_strip<4>(xc16, E_, WcatT, E_, E_, m0, 0, lane, acc);
  int n = lane & 15, hi = lane >> 4;
  float A    = -softplusf(A_log[n]);
  float Ainv = 1.f / (A + 1e-6f);
  float bBn = bB[n], bCn = bC[n], bDn = bD[n], bin = bi[n];
#pragma unroll
  for (int r = 0; r < 8; ++r) {
    int row = m0 + r + hi * 8;
    float Bt  = acc[0][r] + bBn;
    float Ctv = acc[1][r] + bCn;
    float Dt  = softplusf(acc[2][r] + bDn);
    float u   = acc[3][r] + bin;
    float ab  = expf(Dt * A);
    float bu  = (ab - 1.f) * Ainv * Bt * u;
    size_t o = (size_t)row * N_ + n;
    Abar[o] = ab; Bu[o] = bu; Ct[o] = Ctv;
  }
}

// ---- wave-parallel chunked scan: one wave per (b,n) recurrence -------------
__global__ void scan_kernel(const float* __restrict__ Abar, const float* __restrict__ Bu,
                            const float* __restrict__ Ct, _Float16* __restrict__ hsCt16,
                            int dir) {
  int wave = threadIdx.x >> 5, lane = threadIdx.x & 31;
  int pair = blockIdx.x * (blockDim.x >> 5) + wave;   // 512 pairs
  int b = pair >> 4, n = pair & 15;
  const float* Ab = Abar + ((size_t)b * J_) * N_ + n;
  const float* Ub = Bu   + ((size_t)b * J_) * N_ + n;
  const float* Cb = Ct   + ((size_t)b * J_) * N_ + n;
  int sBase = lane * 32;                               // 32-step chunk per lane
  // pass 1: chunk -> affine transform (aP, bP):  h_out = aP*h_in + bP
  float aP = 1.f, bP = 0.f;
#pragma unroll 4
  for (int i = 0; i < 32; ++i) {
    float a = Ab[(size_t)(sBase + i) * N_];
    float u = Ub[(size_t)(sBase + i) * N_];
    bP = a * bP + u;
    aP *= a;
  }
  // wave inclusive composition scan
  for (int off = 1; off < 32; off <<= 1) {
    float ap = __shfl_up(aP, off, 32);
    float bp = __shfl_up(bP, off, 32);
    if (lane >= off) { bP = aP * bp + bP; aP *= ap; }
  }
  float h = __shfl_up(bP, 1, 32);                     // exclusive prefix = h_in
  if (lane == 0) h = 0.f;
  // pass 2: materialize states, multiply by Ct, write (flip for bwd dir)
#pragma unroll 4
  for (int i = 0; i < 32; ++i) {
    int s = sBase + i;
    float a = Ab[(size_t)s * N_];
    float u = Ub[(size_t)s * N_];
    h = a * h + u;
    float c = Cb[(size_t)s * N_];
    int j = dir ? (J_ - 1 - s) : s;
    hsCt16[((size_t)b * J_ + j) * 32 + dir * 16 + n] = (_Float16)(h * c);
  }
}

// ---- readout GEMM: concat(hsCt_f, hsCt_b) @ [Wr;Wr], fused gate ------------
__global__ void gemm_readout_kernel(const _Float16* __restrict__ hsCt16,
                                    const _Float16* __restrict__ Wr2T,
                                    const float* __restrict__ br,
                                    const _Float16* __restrict__ gate16,
                                    _Float16* __restrict__ ygate16) {
  int lane = threadIdx.x;
  int m0 = blockIdx.x * 16, n0 = blockIdx.y * 64;
  v8f acc[4] = {};
  gemm_strip<4>(hsCt16, 32, Wr2T, 32, 32, m0, n0, lane, acc);
  int nc = lane & 15, hi = lane >> 4;
#pragma unroll
  for (int t = 0; t < 4; ++t) {
    int col = n0 + t * 16 + nc;
    float bb = 2.f * br[col];                          // br added in both dirs
#pragma unroll
    for (int r = 0; r < 8; ++r) {
      int row = m0 + r + hi * 8;
      float y = acc[t][r] + bb;
      float g = (float)gate16[(size_t)row * E_ + col];
      ygate16[(size_t)row * E_ + col] = (_Float16)(y * g);
    }
  }
}

// ---- output GEMM with residual ---------------------------------------------
__global__ void gemm_out_kernel(const _Float16* __restrict__ ygate16,
                                const _Float16* __restrict__ WoT,
                                const float* __restrict__ bo,
                                const float* __restrict__ tok,
                                float* __restrict__ out) {
  int lane = threadIdx.x;
  int m0 = blockIdx.x * 16, n0 = blockIdx.y * 64;
  v8f acc[4] = {};
  gemm_strip<4>(ygate16, E_, WoT, E_, E_, m0, n0, lane, acc);
  int nc = lane & 15, hi = lane >> 4;
#pragma unroll
  for (int t = 0; t < 4; ++t) {
    int col = n0 + t * 16 + nc;
    float bb = bo[col];
#pragma unroll
    for (int r = 0; r < 8; ++r) {
      int row = m0 + r + hi * 8;
      size_t o = (size_t)row * D_ + col;
      out[o] = tok[o] + acc[t][r] + bb;
    }
  }
}

// ---------------------------------------------------------------------------
extern "C" void kernel_launch(void* const* d_in, const int* in_sizes, int n_in,
                              void* d_out, int out_size, void* d_ws, size_t ws_size,
                              hipStream_t stream) {
  const float* tokens = (const float*)d_in[0];
  const float* norm_g = (const float*)d_in[1];
  const float* norm_b = (const float*)d_in[2];
  const float* Wx  = (const float*)d_in[3];  const float* bx = (const float*)d_in[4];
  const float* Wz  = (const float*)d_in[5];  const float* bz = (const float*)d_in[6];
  const float* cwf = (const float*)d_in[7];  const float* cbf = (const float*)d_in[8];
  const float* cwb = (const float*)d_in[9];  const float* cbb = (const float*)d_in[10];
  const float* WBf = (const float*)d_in[11]; const float* bBf = (const float*)d_in[12];
  const float* WCf = (const float*)d_in[13]; const float* bCf = (const float*)d_in[14];
  const float* WDf = (const float*)d_in[15]; const float* bDf = (const float*)d_in[16];
  const float* WBb = (const float*)d_in[17]; const float* bBb = (const float*)d_in[18];
  const float* WCb = (const float*)d_in[19]; const float* bCb = (const float*)d_in[20];
  const float* WDb = (const float*)d_in[21]; const float* bDb = (const float*)d_in[22];
  const float* A_log = (const float*)d_in[23];
  const float* Wi  = (const float*)d_in[24]; const float* bi = (const float*)d_in[25];
  const float* Wr  = (const float*)d_in[26]; const float* br = (const float*)d_in[27];
  const float* Wo  = (const float*)d_in[28]; const float* bo = (const float*)d_in[29];
  float* out = (float*)d_out;

  // workspace carve-up (256B aligned)
  char* ws = (char*)d_ws;
  size_t off = 0;
  auto alloc = [&](size_t bytes) -> char* {
    char* p = ws + off;
    off = (off + bytes + 255) & ~(size_t)255;
    return p;
  };
  _Float16* t16     = (_Float16*)alloc((size_t)M_ * D_ * 2);
  _Float16* WxzT    = (_Float16*)alloc((size_t)768 * D_ * 2);
  _Float16* WcatTf  = (_Float16*)alloc((size_t)64 * E_ * 2);
  _Float16* WcatTb  = (_Float16*)alloc((size_t)64 * E_ * 2);
  _Float16* Wr2T    = (_Float16*)alloc((size_t)E_ * 32 * 2);
  _Float16* WoT     = (_Float16*)alloc((size_t)D_ * E_ * 2);
  _Float16* x16     = (_Float16*)alloc((size_t)M_ * E_ * 2);
  _Float16* gate16  = (_Float16*)alloc((size_t)M_ * E_ * 2);
  _Float16* xc16    = (_Float16*)alloc((size_t)M_ * E_ * 2);
  float*    AbarB   = (float*)alloc((size_t)M_ * N_ * 4);
  float*    BuB     = (float*)alloc((size_t)M_ * N_ * 4);
  float*    CtB     = (float*)alloc((size_t)M_ * N_ * 4);
  _Float16* hsCt16  = (_Float16*)alloc((size_t)M_ * 32 * 2);
  _Float16* ygate16 = (_Float16*)alloc((size_t)M_ * E_ * 2);

  // weight prep
  prep_wxz<<<(768 * D_ + 255) / 256, 256, 0, stream>>>(Wx, Wz, WxzT);
  prep_wcat<<<(64 * E_ + 255) / 256, 256, 0, stream>>>(WBf, WCf, WDf, Wi, WcatTf);
  prep_wcat<<<(64 * E_ + 255) / 256, 256, 0, stream>>>(WBb, WCb, WDb, Wi, WcatTb);
  prep_wr2<<<(E_ * 32 + 255) / 256, 256, 0, stream>>>(Wr, Wr2T);
  prep_woT<<<(D_ * E_ + 255) / 256, 256, 0, stream>>>(Wo, WoT);

  // layernorm (one wave per row)
  ln_kernel<<<M_ / 4, 128, 0, stream>>>(tokens, norm_g, norm_b, t16);

  // x/z projections + gate
  gemm_xz_kernel<<<dim3(M_ / 16, 768 / 64), 32, 0, stream>>>(t16, WxzT, bx, bz,
                                                             x16, gate16);

  // two SSM directions (sequential, reusing conv/proj buffers)
  for (int dir = 0; dir < 2; ++dir) {
    const float* cw = dir ? cwb : cwf;
    const float* cb = dir ? cbb : cbf;
    const _Float16* Wcat = dir ? WcatTb : WcatTf;
    const float* bB = dir ? bBb : bBf;
    const float* bC = dir ? bCb : bCf;
    const float* bD = dir ? bDb : bDf;
    dwconv_kernel<<<(size_t)M_ * E_ / 256, 256, 0, stream>>>(x16, cw, cb, xc16, dir);
    gemm_proj_kernel<<<dim3(M_ / 16, 1), 32, 0, stream>>>(xc16, Wcat, bB, bC, bD, bi,
                                                          A_log, AbarB, BuB, CtB);
    scan_kernel<<<B_ * N_ / 8, 256, 0, stream>>>(AbarB, BuB, CtB, hsCt16, dir);
  }

  // readout (both directions in one K=32 GEMM) with gate fused
  gemm_readout_kernel<<<dim3(M_ / 16, E_ / 64), 32, 0, stream>>>(hsCt16, Wr2T, br,
                                                                 gate16, ygate16);
  // output projection + residual
  gemm_out_kernel<<<dim3(M_ / 16, D_ / 64), 32, 0, stream>>>(ygate16, WoT, bo,
                                                             tokens, out);
}
